// BahdanauAttention_39307540693917
// MI455X (gfx1250) — matrix operand order repaired and checked by reading, not compiled, not run on hardware
//
#include <hip/hip_runtime.h>
#include <hip/hip_bf16.h>
#include <math.h>

// Problem dims (fixed by reference)
#define BB 8
#define TX 128
#define TY 128
#define DD 512
#define HH 512
#define LN_EPS 1e-3f

typedef __attribute__((ext_vector_type(16))) __bf16 v16bf;
typedef __attribute__((ext_vector_type(8)))  float  v8f;

#if __has_builtin(__builtin_amdgcn_tanhf)
#define FAST_TANH(x) __builtin_amdgcn_tanhf(x)
#else
#define FAST_TANH(x) tanhf(x)
#endif

// ---- gfx1250 async global->LDS path (guarded: falls back to plain staging) ----
#if defined(__AMDGCN__) && \
    __has_builtin(__builtin_amdgcn_global_load_async_to_lds_b128) && \
    __has_builtin(__builtin_amdgcn_s_wait_asynccnt)
#define HAVE_ASYNC_LDS 1
typedef __attribute__((__vector_size__(4 * sizeof(int)))) int async_b128_t;
#define ASYNC_COPY_B128(gsrc, ldst)                                        \
  __builtin_amdgcn_global_load_async_to_lds_b128(                          \
      (__attribute__((address_space(1))) async_b128_t*)(gsrc),             \
      (__attribute__((address_space(3))) async_b128_t*)(ldst), 0, 0)
#define ASYNC_WAIT(n) __builtin_amdgcn_s_wait_asynccnt(n)
#else
#define HAVE_ASYNC_LDS 0
#define ASYNC_WAIT(n)
#endif

// ---------------------------------------------------------------------------
// Kernel 1: Out[M,H] = A[M,D] @ W[D,H] + bias   (M = B*T = 1024, D=H=512)
// One wave32 per 16x16 output tile, bf16 WMMA, K-step 32, unrolled x4.
// ---------------------------------------------------------------------------
__global__ __launch_bounds__(32)
void proj_gemm_kernel(const float* __restrict__ A,
                      const float* __restrict__ W,
                      const float* __restrict__ bias,
                      float* __restrict__ Out) {
  const int lane = threadIdx.x;
  const int lo = lane & 15;
  const int hi = lane >> 4;

  const int arow = blockIdx.x * 16 + lo;   // A-matrix row for this lane
  const int col  = blockIdx.y * 16 + lo;   // B column / C column for this lane

  const float* __restrict__ ap = A + (size_t)arow * DD;

  v8f acc = {};
#pragma unroll 4
  for (int k0 = 0; k0 < DD; k0 += 32) {
    v16bf av, bv;
    const float* __restrict__ pa0 = ap + k0 + hi * 8;        // K = k0 + hi*8 + j
    const float* __restrict__ pa1 = ap + k0 + 16 + hi * 8;   // K = k0 + 16 + hi*8 + j
#pragma unroll
    for (int j = 0; j < 8; ++j) {
      av[j]     = (__bf16)pa0[j];
      av[8 + j] = (__bf16)pa1[j];
    }
    const int kb = k0 + hi * 8;
#pragma unroll
    for (int j = 0; j < 16; ++j) {
      bv[j] = (__bf16)W[(size_t)(kb + (j & 7) + (j >> 3) * 16) * HH + col];
    }
    acc = __builtin_amdgcn_wmma_f32_16x16x32_bf16(
        /*neg_a=*/false, av, /*neg_b=*/false, bv,
        /*c_mod=*/(short)0, acc, /*reuse_a=*/false, /*reuse_b=*/false);
  }

  const float bc = bias[col];
#pragma unroll
  for (int r = 0; r < 8; ++r) {
    const int m = blockIdx.x * 16 + r + 8 * hi;
    Out[(size_t)m * HH + col] = acc[r] + bc;
  }
}

// ---------------------------------------------------------------------------
// Kernel 2: per (b, ty) row — scores (tanh MLP) with async double-buffered
// Wc tiles in LDS, softmax, context vector, LayerNorm, residual.
// 256 threads = 8 wave32s per block.
// ---------------------------------------------------------------------------
#define CHUNK_ROWS 16
#define NCHUNKS (TX / CHUNK_ROWS)                 // 8
#define CHUNK_FLOATS (CHUNK_ROWS * HH)            // 8192 floats = 32 KB

__global__ __launch_bounds__(256)
void attn_fused_kernel(const float* __restrict__ context,
                       const float* __restrict__ x,
                       const float* __restrict__ Wc,
                       const float* __restrict__ Ux,
                       const float* __restrict__ Va,
                       const float* __restrict__ bVa,
                       const float* __restrict__ gamma,
                       const float* __restrict__ beta,
                       float* __restrict__ out) {
  __shared__ __align__(16) float sWc[2][CHUNK_FLOATS];   // 2 x 32 KB double buffer
  __shared__ float sSc[TX];
  __shared__ float sRedA[8];
  __shared__ float sRedB[8];

  const int t = threadIdx.x;
  const int w = t >> 5;
  const int lane = t & 31;
  const int b  = blockIdx.x / TY;
  const int ty = blockIdx.x % TY;

  const float* __restrict__ wcb = Wc + (size_t)b * TX * HH;

  // Stage chunk c (16 tx-rows of Wc[b]) into LDS buffer `buf`.
  auto issue_copy = [&](int c, int buf) {
#if HAVE_ASYNC_LDS
#pragma unroll
    for (int i = 0; i < 8; ++i) {
      const int fidx = (i * 256 + t) * 4;                  // float4 granularity
      ASYNC_COPY_B128(wcb + (size_t)c * CHUNK_FLOATS + fidx, &sWc[buf][fidx]);
    }
#else
    const float4* __restrict__ src = (const float4*)(wcb + (size_t)c * CHUNK_FLOATS);
    float4* dst = (float4*)sWc[buf];
    for (int i = t; i < CHUNK_FLOATS / 4; i += 256) dst[i] = src[i];
#endif
  };

  // Row-invariant operands live in registers: h = j*32 + lane for all rows.
  float uxr[16], var_[16];
  const float* __restrict__ uxrow = Ux + (size_t)(b * TY + ty) * HH;
#pragma unroll
  for (int j = 0; j < 16; ++j) {
    const int h = j * 32 + lane;
    uxr[j]  = uxrow[h];
    var_[j] = Va[h];
  }
  const float bva = bVa[0];

  // ---- scores with double-buffered async pipeline ----
  issue_copy(0, 0);
  for (int c = 0; c < NCHUNKS; ++c) {
    const int buf = c & 1;
    if (c + 1 < NCHUNKS) {
      issue_copy(c + 1, (c + 1) & 1);
      ASYNC_WAIT(8);          // chunk c's 8 async loads (in-order) are done
    } else {
      ASYNC_WAIT(0);
    }
    __syncthreads();          // chunk c visible block-wide

    // Wave w handles rows 2w, 2w+1 of this 16-row chunk.
#pragma unroll
    for (int rr = 0; rr < 2; ++rr) {
      const int r = w * 2 + rr;
      const float* wrow = &sWc[buf][r * HH];
      float s = 0.f;
#pragma unroll 4
      for (int j = 0; j < 16; ++j) {
        s += var_[j] * FAST_TANH(wrow[j * 32 + lane] + uxr[j]);
      }
#pragma unroll
      for (int off = 16; off > 0; off >>= 1) s += __shfl_xor(s, off, 32);
      if (lane == 0) sSc[c * CHUNK_ROWS + r] = s + bva;
    }
    __syncthreads();          // all waves done with buf before it is refilled
  }

  // ---- softmax over TX=128 (first 4 waves own one score each) ----
  float mval = -__builtin_inff();
  if (t < TX) mval = sSc[t];
#pragma unroll
  for (int off = 16; off > 0; off >>= 1) mval = fmaxf(mval, __shfl_xor(mval, off, 32));
  if (t < TX && lane == 0) sRedA[w] = mval;
  __syncthreads();
  const float gmax = fmaxf(fmaxf(sRedA[0], sRedA[1]), fmaxf(sRedA[2], sRedA[3]));

  float e = 0.f;
  if (t < TX) e = __expf(sSc[t] - gmax);
  float psum = e;
#pragma unroll
  for (int off = 16; off > 0; off >>= 1) psum += __shfl_xor(psum, off, 32);
  if (t < TX && lane == 0) sRedB[w] = psum;
  __syncthreads();
  const float gsum = sRedB[0] + sRedB[1] + sRedB[2] + sRedB[3];
  if (t < TX) sSc[t] = e / gsum;
  __syncthreads();

  // ---- context vector: cv[d] = sum_tx attn[tx] * context[b,tx,d] ----
  const float* __restrict__ ctxb = context + (size_t)b * TX * DD;
  float cv0 = 0.f, cv1 = 0.f;
#pragma unroll 4
  for (int tx = 0; tx < TX; ++tx) {
    const float a = sSc[tx];
    const float* __restrict__ crow = ctxb + (size_t)tx * DD;
    cv0 += a * crow[t];
    cv1 += a * crow[t + 256];
  }

  // ---- LayerNorm over D=512 + residual ----
  float ps = cv0 + cv1;
  float pq = cv0 * cv0 + cv1 * cv1;
#pragma unroll
  for (int off = 16; off > 0; off >>= 1) {
    ps += __shfl_xor(ps, off, 32);
    pq += __shfl_xor(pq, off, 32);
  }
  __syncthreads();   // sRedA/sRedB free for reuse
  if (lane == 0) { sRedA[w] = ps; sRedB[w] = pq; }
  __syncthreads();
  float tot = 0.f, totq = 0.f;
#pragma unroll
  for (int i = 0; i < 8; ++i) { tot += sRedA[i]; totq += sRedB[i]; }
  const float mean = tot * (1.0f / DD);
  const float var  = totq * (1.0f / DD) - mean * mean;
  const float rstd = rsqrtf(var + LN_EPS);

  const size_t rowoff = (size_t)(b * TY + ty) * DD;
  out[rowoff + t]       = (cv0 - mean) * rstd * gamma[t]       + beta[t]       + x[rowoff + t];
  out[rowoff + t + 256] = (cv1 - mean) * rstd * gamma[t + 256] + beta[t + 256] + x[rowoff + t + 256];
}

// ---------------------------------------------------------------------------
// Launch
// ---------------------------------------------------------------------------
extern "C" void kernel_launch(void* const* d_in, const int* in_sizes, int n_in,
                              void* d_out, int out_size, void* d_ws, size_t ws_size,
                              hipStream_t stream) {
  const float* context = (const float*)d_in[0];
  const float* x       = (const float*)d_in[1];
  const float* Wa      = (const float*)d_in[2];
  const float* bWa     = (const float*)d_in[3];
  const float* Ua      = (const float*)d_in[4];
  const float* bUa     = (const float*)d_in[5];
  const float* Va      = (const float*)d_in[6];
  const float* bVa     = (const float*)d_in[7];
  const float* gamma   = (const float*)d_in[8];
  const float* beta    = (const float*)d_in[9];
  float* out = (float*)d_out;

  float* Wc = (float*)d_ws;                        // [B*TX, H]  = 2 MB
  float* Ux = Wc + (size_t)BB * TX * HH;           // [B*TY, H]  = 2 MB

  dim3 g1(BB * TX / 16, HH / 16, 1);
  proj_gemm_kernel<<<g1, 32, 0, stream>>>(context, Wa, bWa, Wc);
  proj_gemm_kernel<<<g1, 32, 0, stream>>>(x, Ua, bUa, Ux);

  attn_fused_kernel<<<BB * TY, 256, 0, stream>>>(context, x, Wc, Ux, Va, bVa,
                                                 gamma, beta, out);
}
